// NPUSafeCrossAttnDecoder2d_85950885528385
// MI455X (gfx1250) — matrix operand order repaired and checked by reading, not compiled, not run on hardware
//
#include <hip/hip_runtime.h>

#define B_  4
#define C_  256
#define T_  128
#define NB_ 64
#define NF_ 512
#define PB_ (T_*NB_)   // 8192
#define PF_ (T_*NF_)   // 65536

typedef __attribute__((ext_vector_type(16))) _Float16 v16h;
typedef __attribute__((ext_vector_type(8)))  _Float16 v8h;
typedef __attribute__((ext_vector_type(8)))  float    v8f;

// ---------------------------------------------------------------------------
// CDNA5 async / TDM capability probes (dormant; never launched).  Toolchain is
// the clang-23/therock lane: tensor_load_to_lds takes 6 args; async-to-LDS
// takes (as1 v4i*, as3 v4i*, i32 offset, i32 cpol).
#if defined(__has_builtin)
#  if __has_builtin(__builtin_amdgcn_tensor_load_to_lds)
#    define HAVE_TDM 1
#  endif
#  if __has_builtin(__builtin_amdgcn_global_load_async_to_lds_b128)
#    define HAVE_ASYNC_LDS 1
#  endif
#endif

#if defined(HAVE_TDM)
typedef unsigned int u32x4 __attribute__((ext_vector_type(4)));
typedef int          i32x4 __attribute__((ext_vector_type(4)));
typedef int          i32x8 __attribute__((ext_vector_type(8)));
__global__ void cdna5_tdm_probe() {
  u32x4 g0 = {0, 0, 0, 0};
  i32x8 g1 = {0, 0, 0, 0, 0, 0, 0, 0};
  i32x4 g2 = {0, 0, 0, 0}, g3 = {0, 0, 0, 0};
  i32x8 g4 = {0, 0, 0, 0, 0, 0, 0, 0};
  __builtin_amdgcn_tensor_load_to_lds(g0, g1, g2, g3, g4, 0);
  __builtin_amdgcn_s_wait_tensorcnt(0);
}
#endif

#if defined(HAVE_ASYNC_LDS)
typedef int v4i_vs __attribute__((vector_size(16)));   // matches builtin pointee
__global__ void cdna5_async_probe(const float* __restrict__ g, float* __restrict__ o) {
  __shared__ float buf[64];
  __builtin_amdgcn_global_load_async_to_lds_b128(
      (__attribute__((address_space(1))) v4i_vs*)g,
      (__attribute__((address_space(3))) v4i_vs*)buf, 0, 0);
#  if __has_builtin(__builtin_amdgcn_s_wait_asynccnt)
  __builtin_amdgcn_s_wait_asynccnt(0);
#  else
  asm volatile("s_wait_asynccnt 0x0" ::: "memory");
#  endif
  __syncthreads();
  o[threadIdx.x] = buf[threadIdx.x & 63];
}
#endif

// ---------------------------------------------------------------------------
// Concatenate two packed 8x f16 chunks into one WMMA fragment register group.
__device__ __forceinline__ v16h frag_cat(v8h lo, v8h hi) {
  return __builtin_shufflevector(lo, hi, 0, 1, 2, 3, 4, 5, 6, 7,
                                         8, 9, 10, 11, 12, 13, 14, 15);
}

// A 16x32 f16 fragment from a row-major matrix (per ISA 7.12.2):
// lane 0-15 row M=lane; elems 0..7 -> K=kh*8+j, elems 8..15 -> K=16+kh*8+j.
__device__ __forceinline__ v16h wmma_a_frag_rowmajor(const _Float16* __restrict__ A, int ld) {
  const int lane = threadIdx.x & 31;
  const int m = lane & 15, kh = lane >> 4;
  const _Float16* p = A + (size_t)m * ld + kh * 8;
  return frag_cat(*(const v8h*)p, *(const v8h*)(p + 16));
}

// B 32x16 f16 fragment from K-contiguous storage (element (K,n) at n*ld + K):
// lane = column n; elems 0..7 -> K = kh*8 + j, elems 8..15 -> K = 16 + kh*8 + j  ->  two 16B loads.
__device__ __forceinline__ v16h wmma_b_frag_kcont(const _Float16* __restrict__ Bt, int ld) {
  const int lane = threadIdx.x & 31;
  const int n = lane & 15, kh = lane >> 4;
  const _Float16* p = Bt + (size_t)n * ld + kh * 8;
  return frag_cat(*(const v8h*)p, *(const v8h*)(p + 16));
}

__device__ __forceinline__ float siluf(float x) { return x / (1.f + expf(-x)); }

// ---------------------------------------------------------------------------
// Generic f16 WMMA GEMM over transposed activations:
//   acc[o][p] = sum_c W[o][c] * XT[b][p][c],  p-tile 64, o-tile 128, K-step 32.
// block = 256 thr = 8 waves; wave computes 16 o-rows x 64 p-cols (4 accums per
// A-fragment).  All fragment loads are contiguous 16B; no LDS, no barriers —
// intra-block B-tile reuse is served by WGP$/L2.
enum { EP_F16 = 0, EP_SILU_F16 = 1, EP_F16_CM = 2, EP_F32_SKIP = 3, EP_F32_ACC = 4 };

template <int EP>
__global__ __launch_bounds__(256) void gemm_f16(
    const _Float16* __restrict__ W, const float* __restrict__ bias,
    const _Float16* __restrict__ XT, void* __restrict__ outv,
    const _Float16* __restrict__ skipT, const float* __restrict__ skip_scale,
    int O, int Cd, int P)
{
  const int b   = blockIdx.z;
  const int p0  = blockIdx.x * 64;
  const int o0  = blockIdx.y * 128;
  const int wave = threadIdx.x >> 5, lane = threadIdx.x & 31;
  const int mw  = o0 + wave * 16;
  const int n = lane & 15, kh = lane >> 4;
  const _Float16* XTb = XT + ((size_t)b * P + p0) * Cd;
  v8f acc[4] = {{}, {}, {}, {}};

  for (int k0 = 0; k0 < Cd; k0 += 32) {
    v16h a = wmma_a_frag_rowmajor(W + (size_t)mw * Cd + k0, Cd);
#pragma unroll
    for (int nt = 0; nt < 4; ++nt) {
      v16h bf = wmma_b_frag_kcont(XTb + (size_t)(nt * 16) * Cd + k0, Cd);
      acc[nt] = __builtin_amdgcn_wmma_f32_16x16x32_f16(false, a, false, bf, (short)0, acc[nt], false, false);
    }
  }

  // C/D layout: vgpr r -> row M = r + kh*8 ; col N = lane&15
  if (EP == EP_F16 || EP == EP_SILU_F16) {
    // transposed store [B][P][O]: lane owns 8 consecutive o -> one 16B store per tile
    _Float16* outT = (_Float16*)outv;
#pragma unroll
    for (int nt = 0; nt < 4; ++nt) {
      v8h h;
#pragma unroll
      for (int r = 0; r < 8; ++r) {
        float v = acc[nt][r] + bias[mw + r + kh * 8];
        h[r] = (_Float16)(EP == EP_SILU_F16 ? siluf(v) : v);
      }
      *(v8h*)(outT + ((size_t)b * P + p0 + nt * 16 + n) * O + mw + kh * 8) = h;
    }
  } else if (EP == EP_F16_CM) {
    // channel-major store [B][O][P] (for V, consumed as an A-matrix later)
    _Float16* out = (_Float16*)outv;
#pragma unroll
    for (int nt = 0; nt < 4; ++nt) {
      const size_t pcol = (size_t)p0 + nt * 16 + n;
#pragma unroll
      for (int r = 0; r < 8; ++r) {
        const int o = mw + r + kh * 8;
        out[(size_t)b * O * P + (size_t)o * P + pcol] = (_Float16)(acc[nt][r] + bias[o]);
      }
    }
  } else if (EP == EP_F32_SKIP) {
    // hidden = acc + bias + ss*query_h  -> channel-major f32 d_out
    float* out = (float*)outv;
    const float ss = *skip_scale;
#pragma unroll
    for (int nt = 0; nt < 4; ++nt) {
      const size_t pcol = (size_t)p0 + nt * 16 + n;
      v8h sk = *(const v8h*)(skipT + ((size_t)b * P + pcol) * Cd + mw + kh * 8);
#pragma unroll
      for (int r = 0; r < 8; ++r) {
        const int o = mw + r + kh * 8;
        out[(size_t)b * O * P + (size_t)o * P + pcol] = acc[nt][r] + bias[o] + ss * (float)sk[r];
      }
    }
  } else {
    // accumulate into channel-major f32 d_out
    float* out = (float*)outv;
#pragma unroll
    for (int nt = 0; nt < 4; ++nt) {
      const size_t pcol = (size_t)p0 + nt * 16 + n;
#pragma unroll
      for (int r = 0; r < 8; ++r) {
        const int o = mw + r + kh * 8;
        out[(size_t)b * O * P + (size_t)o * P + pcol] += acc[nt][r] + bias[o];
      }
    }
  }
}

// ---------------------------------------------------------------------------
// RMSNorm over channel dim; f32 channel-major in -> f16 TRANSPOSED out [B][P][C]
__global__ void rms_to_f16_T(const float* __restrict__ x, const float* __restrict__ w,
                             _Float16* __restrict__ yT, int Cd, int P)
{
  const int b = blockIdx.y;
  const int p = blockIdx.x * blockDim.x + threadIdx.x;
  if (p >= P) return;
  const float* xb = x + (size_t)b * Cd * P + p;
  float s = 0.f;
  for (int c = 0; c < Cd; ++c) { float v = xb[(size_t)c * P]; s += v * v; }
  const float inv = rsqrtf(s / (float)Cd + 1e-6f);
  _Float16* yp = yT + ((size_t)b * P + p) * Cd;
  for (int c = 0; c < Cd; ++c) yp[c] = (_Float16)(xb[(size_t)c * P] * inv * w[c]);
}

// transposed swiglu: oT[b][p][c] = a * silu(g), a = hT[b][p][c], g = hT[b][p][C+c]
__global__ void swiglu_mul_T(const _Float16* __restrict__ hT, _Float16* __restrict__ oT,
                             int Cd, int P)
{
  const int b = blockIdx.y;
  const size_t i = (size_t)blockIdx.x * blockDim.x + threadIdx.x;  // over P * Cd/8
  const size_t nch = (size_t)Cd / 8;
  if (i >= (size_t)P * nch) return;
  const size_t p = i / nch, c8 = (i % nch) * 8;
  const _Float16* hp = hT + ((size_t)b * P + p) * 2 * Cd;
  v8h a = *(const v8h*)(hp + c8);
  v8h g = *(const v8h*)(hp + Cd + c8);
  v8h o;
#pragma unroll
  for (int j = 0; j < 8; ++j) {
    float gf = (float)g[j];
    o[j] = (_Float16)((float)a[j] * siluf(gf));
  }
  *(v8h*)(oT + ((size_t)b * P + p) * Cd + c8) = o;
}

__global__ void f32_to_f16(const float* __restrict__ in, _Float16* __restrict__ out, int n)
{
  int i = blockIdx.x * blockDim.x + threadIdx.x;
  if (i < n) out[i] = (_Float16)in[i];
}

// ---------------------------------------------------------------------------
// scores[b,t,f,n] = (q . k)*sc + basis[n,f]*pr ; softmax over n (NB=64) in
// registers via 16-lane butterflies; store weights f16 [bt][f][n].
// qT [B][T*NF][C], kT [B][T*NB][C] -> all fragment loads contiguous 16B.
// grid: (NF/128, B*T), block 256; wave owns 16 f-rows x 64 n (4 accumulators).
__global__ __launch_bounds__(256) void attn_scores(
    const _Float16* __restrict__ qT, const _Float16* __restrict__ kT,
    const float* __restrict__ basis, const float* __restrict__ score_scale,
    const float* __restrict__ prior_scale, _Float16* __restrict__ wt)
{
  const int bt = blockIdx.y;
  const int bb = bt / T_, t = bt % T_;
  const int wave = threadIdx.x >> 5, lane = threadIdx.x & 31;
  const int f0 = blockIdx.x * 128 + wave * 16;
  const int m = lane & 15, kh = lane >> 4;
  const _Float16* qTb = qT + (size_t)bb * PF_ * C_ + (size_t)(t * NF_ + f0 + m) * C_;
  const _Float16* kTb = kT + (size_t)bb * PB_ * C_ + (size_t)(t * NB_) * C_;
  v8f acc[4] = {{}, {}, {}, {}};

  for (int k0 = 0; k0 < C_; k0 += 32) {
    const _Float16* pA = qTb + k0 + kh * 8;
    v16h a = frag_cat(*(const v8h*)pA, *(const v8h*)(pA + 16));
#pragma unroll
    for (int nt = 0; nt < 4; ++nt) {
      v16h bf = wmma_b_frag_kcont(kTb + (size_t)(nt * 16) * C_ + k0, C_);
      acc[nt] = __builtin_amdgcn_wmma_f32_16x16x32_f16(false, a, false, bf, (short)0, acc[nt], false, false);
    }
  }

  const float sc = *score_scale, pr = *prior_scale;
  float vals[4][8];
#pragma unroll
  for (int nt = 0; nt < 4; ++nt)
#pragma unroll
    for (int r = 0; r < 8; ++r) {
      const int f = f0 + r + kh * 8;
      const int n = nt * 16 + m;
      vals[nt][r] = acc[nt][r] * sc + pr * basis[(size_t)n * NF_ + f];
    }
#pragma unroll
  for (int r = 0; r < 8; ++r) {
    float mx = fmaxf(fmaxf(vals[0][r], vals[1][r]), fmaxf(vals[2][r], vals[3][r]));
#pragma unroll
    for (int d = 1; d < 16; d <<= 1) mx = fmaxf(mx, __shfl_xor(mx, d, 32));
    float sum = 0.f;
#pragma unroll
    for (int nt = 0; nt < 4; ++nt) { vals[nt][r] = expf(vals[nt][r] - mx); sum += vals[nt][r]; }
#pragma unroll
    for (int d = 1; d < 16; d <<= 1) sum += __shfl_xor(sum, d, 32);
    const float inv = 1.f / sum;
    const int f = f0 + r + kh * 8;
#pragma unroll
    for (int nt = 0; nt < 4; ++nt)
      wt[((size_t)bt * NF_ + f) * NB_ + nt * 16 + m] = (_Float16)(vals[nt][r] * inv);
  }
}

// attendedT[b][t*NF+f][c] = sum_n wt[t,f,n] * v[c,t,n] ; per (b,t) GEMM, K=NB=64.
// A from channel-major v (contiguous n), B from wt (contiguous n), packed
// transposed stores (lane owns 8 consecutive c).  grid: (NF/64, C/128, B*T).
__global__ __launch_bounds__(256) void attn_av(
    const _Float16* __restrict__ v, const _Float16* __restrict__ wt,
    _Float16* __restrict__ attT)
{
  const int bt = blockIdx.z, bb = bt / T_, t = bt % T_;
  const int wave = threadIdx.x >> 5, lane = threadIdx.x & 31;
  const int c0 = blockIdx.y * 128 + wave * 16;
  const int fblk = blockIdx.x * 64;
  const int m = lane & 15, kh = lane >> 4;
  const _Float16* vb = v + (size_t)bb * C_ * PB_ + (size_t)t * NB_;
  const _Float16* wb = wt + (size_t)bt * NF_ * NB_;
  v8f acc[4] = {{}, {}, {}, {}};

#pragma unroll
  for (int kk = 0; kk < 2; ++kk) {
    const int n0 = kk * 32;
    const _Float16* pA = vb + (size_t)(c0 + m) * PB_ + n0 + kh * 8;
    v16h a = frag_cat(*(const v8h*)pA, *(const v8h*)(pA + 16));
#pragma unroll
    for (int nt = 0; nt < 4; ++nt) {
      v16h bf = wmma_b_frag_kcont(wb + (size_t)(fblk + nt * 16) * NB_ + n0, NB_);
      acc[nt] = __builtin_amdgcn_wmma_f32_16x16x32_f16(false, a, false, bf, (short)0, acc[nt], false, false);
    }
  }
  _Float16* ab = attT + ((size_t)bb * PF_ + (size_t)t * NF_) * C_;
#pragma unroll
  for (int nt = 0; nt < 4; ++nt) {
    v8h h;
#pragma unroll
    for (int r = 0; r < 8; ++r) h[r] = (_Float16)acc[nt][r];
    *(v8h*)(ab + (size_t)(fblk + nt * 16 + m) * C_ + c0 + kh * 8) = h;
  }
}

// ---------------------------------------------------------------------------
extern "C" void kernel_launch(void* const* d_in, const int* in_sizes, int n_in,
                              void* d_out, int out_size, void* d_ws, size_t ws_size,
                              hipStream_t stream)
{
  (void)in_sizes; (void)n_in; (void)out_size; (void)ws_size;
  const float* latent = (const float*)d_in[0];
  const float* side   = (const float*)d_in[1];
  const float* basis  = (const float*)d_in[2];
  const float* lp_nw  = (const float*)d_in[3];
  const float* lp_w   = (const float*)d_in[4];
  const float* lp_b   = (const float*)d_in[5];
  const float* qn_w   = (const float*)d_in[6];
  const float* qi_w   = (const float*)d_in[7];
  const float* qi_b   = (const float*)d_in[8];
  const float* qo_w   = (const float*)d_in[9];
  const float* qo_b   = (const float*)d_in[10];
  const float* q_w    = (const float*)d_in[11];
  const float* q_b    = (const float*)d_in[12];
  const float* k_w    = (const float*)d_in[13];
  const float* k_b    = (const float*)d_in[14];
  const float* v_w    = (const float*)d_in[15];
  const float* v_b    = (const float*)d_in[16];
  const float* o_w    = (const float*)d_in[17];
  const float* o_b    = (const float*)d_in[18];
  const float* fn_w   = (const float*)d_in[19];
  const float* fi_w   = (const float*)d_in[20];
  const float* fi_b   = (const float*)d_in[21];
  const float* fo_w   = (const float*)d_in[22];
  const float* fo_b   = (const float*)d_in[23];
  const float* s_scale   = (const float*)d_in[24];
  const float* p_scale   = (const float*)d_in[25];
  const float* skip_sc   = (const float*)d_in[26];
  float* out = (float*)d_out;

  // ---- workspace carve (all 256B aligned) ----
  char* p = (char*)d_ws;
  auto carve = [&](size_t bytes) { char* r = p; p += (bytes + 255) & ~(size_t)255; return r; };
  const size_t CC = (size_t)C_ * C_, C2C = 2 * CC;
  _Float16* w_lp = (_Float16*)carve(CC * 2);
  _Float16* w_qi = (_Float16*)carve(C2C * 2);
  _Float16* w_qo = (_Float16*)carve(CC * 2);
  _Float16* w_q  = (_Float16*)carve(CC * 2);
  _Float16* w_k  = (_Float16*)carve(CC * 2);
  _Float16* w_v  = (_Float16*)carve(CC * 2);
  _Float16* w_o  = (_Float16*)carve(CC * 2);
  _Float16* w_fi = (_Float16*)carve(C2C * 2);
  _Float16* w_fo = (_Float16*)carve(CC * 2);
  const size_t SF = (size_t)B_ * C_ * PF_;
  const size_t SB = (size_t)B_ * C_ * PB_;
  _Float16* R0 = (_Float16*)carve(SF * 2);        // side_n^T -> hidden_n^T
  _Float16* R1 = (_Float16*)carve(2 * SF * 2);    // qmlp hidden^T -> ffn hidden^T
  _Float16* R2 = (_Float16*)carve(SF * 2);        // swiglu mid^T (both MLPs)
  _Float16* R3 = (_Float16*)carve(SF * 2);        // query_h^T (kept for skip)
  _Float16* R4 = (_Float16*)carve(SF * 2);        // qT -> attended^T
  _Float16* R5 = (_Float16*)carve(SB * 2);        // latent_n^T
  _Float16* R6 = (_Float16*)carve(SB * 2);        // latent_h^T
  _Float16* R7 = (_Float16*)carve(SB * 2);        // kT  [B][PB][C]
  _Float16* R8 = (_Float16*)carve(SB * 2);        // v   [B][C][PB] (channel-major)
  _Float16* R9 = (_Float16*)carve((size_t)B_ * T_ * NF_ * NB_ * 2); // softmax weights

  auto cvt = [&](const float* s, _Float16* d, int nelem) {
    f32_to_f16<<<dim3((nelem + 255) / 256), 256, 0, stream>>>(s, d, nelem);
  };
  cvt(lp_w, w_lp, (int)CC); cvt(qi_w, w_qi, (int)C2C); cvt(qo_w, w_qo, (int)CC);
  cvt(q_w,  w_q,  (int)CC); cvt(k_w,  w_k,  (int)CC);  cvt(v_w,  w_v,  (int)CC);
  cvt(o_w,  w_o,  (int)CC); cvt(fi_w, w_fi, (int)C2C); cvt(fo_w, w_fo, (int)CC);

  // latent path: rms -> silu(conv) -> kT (transposed), v (channel-major)
  rms_to_f16_T<<<dim3(PB_ / 256, B_), 256, 0, stream>>>(latent, lp_nw, R5, C_, PB_);
  gemm_f16<EP_SILU_F16><<<dim3(PB_/64, C_/128, B_), 256, 0, stream>>>(w_lp, lp_b, R5, R6, nullptr, nullptr, C_, C_, PB_);
  gemm_f16<EP_F16><<<dim3(PB_/64, C_/128, B_), 256, 0, stream>>>(w_k, k_b, R6, R7, nullptr, nullptr, C_, C_, PB_);
  gemm_f16<EP_F16_CM><<<dim3(PB_/64, C_/128, B_), 256, 0, stream>>>(w_v, v_b, R6, R8, nullptr, nullptr, C_, C_, PB_);

  // query path: rms -> swiglu MLP -> qT
  rms_to_f16_T<<<dim3(PF_ / 256, B_), 256, 0, stream>>>(side, qn_w, R0, C_, PF_);
  gemm_f16<EP_F16><<<dim3(PF_/64, (2*C_)/128, B_), 256, 0, stream>>>(w_qi, qi_b, R0, R1, nullptr, nullptr, 2*C_, C_, PF_);
  swiglu_mul_T<<<dim3((unsigned)(((size_t)PF_ * (C_/8) + 255) / 256), B_), 256, 0, stream>>>(R1, R2, C_, PF_);
  gemm_f16<EP_F16><<<dim3(PF_/64, C_/128, B_), 256, 0, stream>>>(w_qo, qo_b, R2, R3, nullptr, nullptr, C_, C_, PF_);
  gemm_f16<EP_F16><<<dim3(PF_/64, C_/128, B_), 256, 0, stream>>>(w_q, q_b, R3, R4, nullptr, nullptr, C_, C_, PF_);

  // attention
  attn_scores<<<dim3(NF_/128, B_*T_), 256, 0, stream>>>(R4, R7, basis, s_scale, p_scale, R9);
  attn_av<<<dim3(NF_/64, C_/128, B_*T_), 256, 0, stream>>>(R8, R9, R4);
  gemm_f16<EP_F32_SKIP><<<dim3(PF_/64, C_/128, B_), 256, 0, stream>>>(w_o, o_b, R4, out, R3, skip_sc, C_, C_, PF_);

  // FFN: rms -> swiglu MLP, accumulated into d_out
  rms_to_f16_T<<<dim3(PF_ / 256, B_), 256, 0, stream>>>(out, fn_w, R0, C_, PF_);
  gemm_f16<EP_F16><<<dim3(PF_/64, (2*C_)/128, B_), 256, 0, stream>>>(w_fi, fi_b, R0, R1, nullptr, nullptr, 2*C_, C_, PF_);
  swiglu_mul_T<<<dim3((unsigned)(((size_t)PF_ * (C_/8) + 255) / 256), B_), 256, 0, stream>>>(R1, R2, C_, PF_);
  gemm_f16<EP_F32_ACC><<<dim3(PF_/64, C_/128, B_), 256, 0, stream>>>(w_fo, fo_b, R2, out, nullptr, nullptr, C_, C_, PF_);
}
